// DyDySample_89111981458017
// MI455X (gfx1250) — matrix operand order repaired and checked by reading, not compiled, not run on hardware
//
#include <hip/hip_runtime.h>
#include <cstdint>

#define DEVI __device__ __forceinline__

typedef _Float16 half_t;
typedef _Float16 v16h __attribute__((ext_vector_type(16)));
typedef float    v8f  __attribute__((ext_vector_type(8)));

static constexpr int Bn = 8, Cc = 128, Hh = 128, Ww = 128;
static constexpr int Gg = 4, Ss = 2;
static constexpr int OH = Hh * Ss, OW = Ww * Ss;

// LDS strides (in elements) chosen for 16B alignment + bank-conflict avoidance
static constexpr int ACT_STRIDE = 136;  // halves per pixel row (272 B, 16B-aligned, bank stride 68 dw)
static constexpr int OUT_STRIDE = 132;  // floats per pixel row (bank stride 4)
static constexpr int TR_STRIDE  = 132;  // halves, for the NCHW->NHWC transpose tile

union Frag16 { v16h v; uint32_t u[8]; uint4 q[2]; };

// ---- workspace layout ----
static constexpr size_t XH_OFF    = 0;                                   // f16 NHWC copy of x
static constexpr size_t XH_BYTES  = (size_t)Bn * Hh * Ww * Cc * 2;       // 33,554,432
static constexpr size_t CWS_OFF   = XH_OFF + XH_BYTES;                   // f32 coords, 32 per input px
static constexpr size_t CWS_BYTES = (size_t)Bn * Hh * Ww * 32 * 4;       // 16,777,216
static constexpr size_t EWF_OFF   = CWS_OFF + CWS_BYTES;                 // end_w WMMA-B frags (32 x 512 halves)
static constexpr size_t EWF_BYTES = (size_t)32 * 512 * 2;
static constexpr size_t OWF_OFF   = EWF_OFF + EWF_BYTES;                 // offset+scope WMMA-B frags (16 x 512)
static constexpr size_t OWF_BYTES = (size_t)16 * 512 * 2;
static constexpr size_t WS_NEED   = OWF_OFF + OWF_BYTES;

DEVI uint32_t pack2h(float a, float b) {
    union { half_t h[2]; uint32_t u; } p;
    p.h[0] = (half_t)a; p.h[1] = (half_t)b;
    return p.u;
}

// Load one 16x32 f16 A-fragment from LDS activations per documented VGPR layout:
// lanes 0-15: m=lane, V0..3 = K(kh*8 + 0..7), V4..7 = K(16 + kh*8 + 0..7); lanes 16-31: kh=1.
DEVI Frag16 load_a_frag(const half_t* act, int wave, int nl, int kh, int kb) {
    Frag16 a;
    const half_t* ap = act + (wave * 16 + nl) * ACT_STRIDE + kb * 32 + kh * 8;
    a.u[0] = *(const uint32_t*)(ap + 0);
    a.u[1] = *(const uint32_t*)(ap + 2);
    a.u[2] = *(const uint32_t*)(ap + 4);
    a.u[3] = *(const uint32_t*)(ap + 6);
    a.u[4] = *(const uint32_t*)(ap + 16);
    a.u[5] = *(const uint32_t*)(ap + 18);
    a.u[6] = *(const uint32_t*)(ap + 20);
    a.u[7] = *(const uint32_t*)(ap + 22);
    return a;
}

// ---------------- Kernel 0: x NCHW f32 -> NHWC f16 (LDS tiled transpose) ----------------
__global__ void k_nchw_to_nhwc(const float* __restrict__ x, half_t* __restrict__ xh) {
    __shared__ half_t lds[64 * TR_STRIDE];
    const int wb = blockIdx.x * 64, h = blockIdx.y, b = blockIdx.z;
    const int t = threadIdx.x;
    const float* xp = x + (size_t)b * Cc * Hh * Ww + (size_t)h * Ww + wb;
    #pragma unroll 4
    for (int i = 0; i < 32; ++i) {               // 64 w x 128 c elements
        int e = t + i * 256;
        int c = e >> 6, w = e & 63;              // lanes -> consecutive w (coalesced reads)
        lds[w * TR_STRIDE + c] = (half_t)xp[(size_t)c * Hh * Ww + w];
    }
    __syncthreads();
    uint32_t* xo = (uint32_t*)(xh + (((size_t)b * Hh + h) * Ww + wb) * Cc);
    #pragma unroll 4
    for (int i = 0; i < 16; ++i) {               // 64 w x 64 dwords (2 halves each)
        int e = t + i * 256;
        int w = e >> 6, c2 = e & 63;             // lanes -> consecutive channels (coalesced writes)
        xo[(size_t)w * 64 + c2] = *(const uint32_t*)&lds[w * TR_STRIDE + c2 * 2];
    }
}

// ---------------- Kernel 1: pre-swizzle weights into WMMA B-fragment layout ----------------
// B-frag (16-bit, 32xK16 per nb): lane = {nl=N col, kh}; VGPR v holds K = kb*32 + kh*8 + (v>>2)*16 + 2(v&3) (+1).
__global__ void k_prep_weights(const float* __restrict__ offw, const float* __restrict__ scpw,
                               const float* __restrict__ endw,
                               half_t* __restrict__ ewf, half_t* __restrict__ owf) {
    const int t = threadIdx.x;
    for (int idx = t; idx < 32 * 512; idx += 256) {      // end_w: 4 kb x 8 nb frags
        int frag = idx >> 9, r = idx & 511;
        int lane = r >> 4, hv = r & 15;
        int v = hv >> 1, lo = hv & 1;
        int kb = frag >> 3, nb = frag & 7;
        int nl = lane & 15, kh = lane >> 4;
        int k = kb * 32 + kh * 8 + (v >> 2) * 16 + 2 * (v & 3) + lo;
        int n = nb * 16 + nl;
        ewf[idx] = (half_t)endw[n * 128 + k];
    }
    for (int idx = t; idx < 16 * 512; idx += 256) {      // [offset_w; scope_w]: 4 kb x 4 nb frags
        int frag = idx >> 9, r = idx & 511;
        int lane = r >> 4, hv = r & 15;
        int v = hv >> 1, lo = hv & 1;
        int kb = frag >> 2, nb = frag & 3;
        int nl = lane & 15, kh = lane >> 4;
        int k = kb * 32 + kh * 8 + (v >> 2) * 16 + 2 * (v & 3) + lo;
        int n = nb * 16 + nl;
        float wv = (n < 32) ? offw[n * 128 + k] : scpw[(n - 32) * 128 + k];
        owf[idx] = (half_t)wv;
    }
}

// ---------------- Kernel 2: offset/scope conv (WMMA) + coordinate field ----------------
// Per block: one (b,h) row of 128 pixels. GEMM M=128 (px), K=128 (ch), N=64 (32 off + 32 scope).
// Grid-sample algebra collapses to ix = w + off_x, iy = h + off_y (border-clamped).
__global__ void k_offsets(const half_t* __restrict__ xh, const half_t* __restrict__ owf,
                          const float* __restrict__ offb, float* __restrict__ cws) {
    __shared__ half_t act[128 * ACT_STRIDE];
    const int h = blockIdx.x, b = blockIdx.y;
    const int t = threadIdx.x;

    const uint4* src = (const uint4*)(xh + (((size_t)b * Hh + h) * Ww) * Cc);
    #pragma unroll
    for (int i = 0; i < 4; ++i) {                 // stage 128 px x 64 ch (16 KB); 8-ch chunks land 16 halves apart
        int id = t + i * 256;
        int px = id >> 3, ch = id & 7;
        *(uint4*)&act[px * ACT_STRIDE + ch * 16] = src[px * 16 + ch];
    }
    __syncthreads();

    const int wave = t >> 5, lane = t & 31;
    const int nl = lane & 15, kh = lane >> 4;
    v8f acc[4];
    #pragma unroll
    for (int nb = 0; nb < 4; ++nb)
        #pragma unroll
        for (int q = 0; q < 8; ++q) acc[nb][q] = 0.0f;

    for (int kb = 0; kb < 4; ++kb) {
        Frag16 a = load_a_frag(act, wave, nl, kh, kb);
        #pragma unroll
        for (int nb = 0; nb < 4; ++nb) {
            Frag16 bw;
            const uint4* wp = (const uint4*)(owf + (size_t)((kb * 4 + nb) * 32 + lane) * 16);
            bw.q[0] = wp[0]; bw.q[1] = wp[1];
            acc[nb] = __builtin_amdgcn_wmma_f32_16x16x32_f16(false, a.v, false, bw.v,
                                                             (short)0, acc[nb], false, false);
        }
    }

    // n in [0,32): offset conv result; pair n+32: scope conv result (same lane/v slot, nb+2)
    #pragma unroll
    for (int nb = 0; nb < 2; ++nb) {
        int n = nb * 16 + nl;
        float bias = offb[n];
        #pragma unroll
        for (int v = 0; v < 8; ++v) {
            int px = wave * 16 + v + 8 * kh;       // D: M = v + 8*kh
            float o = acc[nb][v] + bias;
            float s = 1.0f / (1.0f + __expf(-acc[nb + 2][v]));
            float coord; int slot;
            if (n < 16) {                          // x component; init_pos = (sx?0.25:-0.25)
                float val = o * s * 0.5f + (((n & 1) != 0) ? 0.25f : -0.25f);
                coord = fminf(fmaxf((float)px + val, 0.0f), (float)(Ww - 1));
                slot = n * 2;
            } else {                               // y component; init_pos = (sy?0.25:-0.25)
                float val = o * s * 0.5f + ((((n >> 1) & 1) != 0) ? 0.25f : -0.25f);
                coord = fminf(fmaxf((float)h + val, 0.0f), (float)(Hh - 1));
                slot = (n - 16) * 2 + 1;
            }
            cws[(((size_t)b * Hh + h) * Ww + px) * 32 + slot] = coord;
        }
    }
}

// ---------------- Kernel 3: fused bilinear sampler + end 1x1 conv (WMMA) ----------------
// Per block: 128 output pixels (one oy row segment). Sample 128 ch -> LDS f16 -> 32 WMMAs/wave.
__global__ void k_sample_endconv(const half_t* __restrict__ xh,
                                 const float*  __restrict__ cws,
                                 const half_t* __restrict__ ewf,
                                 const float*  __restrict__ endb,
                                 float* __restrict__ out) {
    extern __shared__ char smem_raw[];
    half_t* act  = (half_t*)smem_raw;              // phase 1: 128*136 halves (34,816 B)
    float*  outs = (float*)smem_raw;               // phase 2 (after barrier): 128*132 f32 (67,584 B)

    const int oxb = blockIdx.x * 128;
    const int oy  = blockIdx.y;
    const int b   = blockIdx.z;
    const int t   = threadIdx.x;
    const int h   = oy >> 1, sy = oy & 1;

    // --- sampling: 128 px x 4 groups x 4 chunks(8ch) = 2048 tasks; lanes 0..15 share taps per group
    for (int i = 0; i < 8; ++i) {
        int id = i * 256 + t;
        int px = id >> 4, sub = id & 15;
        int gi = sub >> 2, ck = sub & 3;
        int ox = oxb + px;
        int w_in = ox >> 1, sx = ox & 1;
        int n = gi * 4 + sy * 2 + sx;
        float2 cxy = *(const float2*)(cws + ((((size_t)b * Hh + h) * Ww + w_in) * 32 + n * 2));
        float ix = cxy.x, iy = cxy.y;
        float x0f = floorf(ix), y0f = floorf(iy);
        float wx = ix - x0f, wy = iy - y0f;
        int x0 = (int)x0f, y0 = (int)y0f;
        int x1 = (x0 + 1 > Ww - 1) ? (Ww - 1) : x0 + 1;
        int y1 = (y0 + 1 > Hh - 1) ? (Hh - 1) : y0 + 1;
        int cb = gi * 32 + ck * 8;
        const half_t* rb = xh + (size_t)b * Hh * Ww * Cc;
        uint4 q00 = *(const uint4*)(rb + ((size_t)y0 * Ww + x0) * Cc + cb);
        uint4 q01 = *(const uint4*)(rb + ((size_t)y0 * Ww + x1) * Cc + cb);
        uint4 q10 = *(const uint4*)(rb + ((size_t)y1 * Ww + x0) * Cc + cb);
        uint4 q11 = *(const uint4*)(rb + ((size_t)y1 * Ww + x1) * Cc + cb);
        const half_t* h00 = (const half_t*)&q00;
        const half_t* h01 = (const half_t*)&q01;
        const half_t* h10 = (const half_t*)&q10;
        const half_t* h11 = (const half_t*)&q11;
        float w00 = (1.f - wx) * (1.f - wy), w01 = wx * (1.f - wy);
        float w10 = (1.f - wx) * wy,         w11 = wx * wy;
        float r[8];
        #pragma unroll
        for (int cc = 0; cc < 8; ++cc)
            r[cc] = (float)h00[cc] * w00 + (float)h01[cc] * w01
                  + (float)h10[cc] * w10 + (float)h11[cc] * w11;
        uint4 pk;
        pk.x = pack2h(r[0], r[1]); pk.y = pack2h(r[2], r[3]);
        pk.z = pack2h(r[4], r[5]); pk.w = pack2h(r[6], r[7]);
        *(uint4*)(act + px * ACT_STRIDE + cb) = pk;
    }
    __syncthreads();

    // --- end conv: per wave 16 px x 128 out-ch, K=128 in 4 steps of 32
    const int wave = t >> 5, lane = t & 31;
    const int nl = lane & 15, kh = lane >> 4;
    v8f acc[8];
    #pragma unroll
    for (int nb = 0; nb < 8; ++nb)
        #pragma unroll
        for (int q = 0; q < 8; ++q) acc[nb][q] = 0.0f;

    for (int kb = 0; kb < 4; ++kb) {
        Frag16 a = load_a_frag(act, wave, nl, kh, kb);
        #pragma unroll
        for (int nb = 0; nb < 8; ++nb) {
            Frag16 bw;                              // weight frags stream from L2-resident ws
            const uint4* wp = (const uint4*)(ewf + (size_t)((kb * 8 + nb) * 32 + lane) * 16);
            bw.q[0] = wp[0]; bw.q[1] = wp[1];
            acc[nb] = __builtin_amdgcn_wmma_f32_16x16x32_f16(false, a.v, false, bw.v,
                                                             (short)0, acc[nb], false, false);
        }
    }
    __syncthreads();                                // all act reads done before LDS reuse

    #pragma unroll
    for (int nb = 0; nb < 8; ++nb) {
        int n = nb * 16 + nl;
        #pragma unroll
        for (int v = 0; v < 8; ++v) {
            int px = wave * 16 + v + 8 * kh;        // D: M = v + 8*kh, N = nb*16 + nl
            outs[px * OUT_STRIDE + n] = acc[nb][v];
        }
    }
    __syncthreads();

    // --- coalesced NCHW stores: each wave writes 512 B rows (32 lanes x float4) per channel
    for (int cc = 0; cc < 16; ++cc) {
        int ch = wave * 16 + cc;
        float bias = endb[ch];
        int px = lane * 4;
        float4 vq;
        vq.x = outs[(px + 0) * OUT_STRIDE + ch] + bias;
        vq.y = outs[(px + 1) * OUT_STRIDE + ch] + bias;
        vq.z = outs[(px + 2) * OUT_STRIDE + ch] + bias;
        vq.w = outs[(px + 3) * OUT_STRIDE + ch] + bias;
        *(float4*)(out + (((size_t)b * Cc + ch) * OH + oy) * OW + oxb + px) = vq;
    }
}

extern "C" void kernel_launch(void* const* d_in, const int* in_sizes, int n_in,
                              void* d_out, int out_size, void* d_ws, size_t ws_size,
                              hipStream_t stream) {
    (void)in_sizes; (void)n_in; (void)out_size;
    const float* x    = (const float*)d_in[0];
    const float* offw = (const float*)d_in[1];
    const float* offb = (const float*)d_in[2];
    const float* scpw = (const float*)d_in[3];
    const float* endw = (const float*)d_in[4];
    const float* endb = (const float*)d_in[5];
    float* out = (float*)d_out;
    if (ws_size < WS_NEED) return;
    char* ws = (char*)d_ws;
    half_t* xh  = (half_t*)(ws + XH_OFF);
    float*  cws = (float*)(ws + CWS_OFF);
    half_t* ewf = (half_t*)(ws + EWF_OFF);
    half_t* owf = (half_t*)(ws + OWF_OFF);

    k_nchw_to_nhwc  <<<dim3(Ww / 64, Hh, Bn), 256, 0, stream>>>(x, xh);
    k_prep_weights  <<<dim3(1), 256, 0, stream>>>(offw, scpw, endw, ewf, owf);
    k_offsets       <<<dim3(Hh, Bn), 256, 0, stream>>>(xh, owf, offb, cws);
    k_sample_endconv<<<dim3(OW / 128, OH, Bn), 256, 128 * OUT_STRIDE * 4, stream>>>(
        xh, cws, ewf, endb, out);
}
